// GNN_Model_20942260536086
// MI455X (gfx1250) — matrix operand order repaired and checked by reading, not compiled, not run on hardware
//
#include <hip/hip_runtime.h>

typedef __attribute__((ext_vector_type(16))) __bf16 v16bf;
typedef __attribute__((ext_vector_type(8)))  float  v8f;

#define N_NODES 10000
#define N_EDGES 320000
#define HID     256
#define NCLS    101
#define NGRAPH  64
#define TROW    264   // LDS row stride in bf16 (256 + 8 pad -> conflict-free frag loads)
#define EPT     32    // edges per wave (two 16-row M-tiles)

union Frag { uint4 u[2]; v16bf v; };

// ---------------------------------------------------------------- init / zero
__global__ void k_init(float* __restrict__ acc, float* __restrict__ pool, int* __restrict__ cnt) {
    int id = blockIdx.x * 256 + threadIdx.x;
    if (id < N_NODES * HID) acc[id] = 0.f;
    if (id < NGRAPH * HID)  pool[id] = 0.f;
    if (id < NGRAPH)        cnt[id] = 0;
}

// --------------------------------------------------- weight pack (fragment-major bf16)
// dst[((kt*16 + nt)*32 + lane)*16 + h] = W[(rowOff + kt*32 + (lane>>4)*8 + (h&7) + (h>>3)*16)*256 + nt*16 + (lane&15)]
__global__ void k_pack(const float* __restrict__ W, int rowOff, __bf16* __restrict__ dst, int total) {
    int id = blockIdx.x * 256 + threadIdx.x;
    if (id >= total) return;
    int h  = id & 15;
    int l  = (id >> 4) & 31;
    int nt = (id >> 9) & 15;
    int kt = id >> 13;
    int k  = kt * 32 + ((l >> 4) * 8) + (h & 7) + ((h >> 3) * 16);
    int n  = nt * 16 + (l & 15);
    dst[id] = (__bf16)W[(size_t)(rowOff + k) * HID + n];
}

// ------------------------------------------------- layer-0 node transform (K=10, scalar)
__global__ void k_node0(const float* __restrict__ x, const float* __restrict__ W1,
                        const float* __restrict__ b1, __bf16* __restrict__ hA,
                        __bf16* __restrict__ hB) {
    int id = blockIdx.x * 256 + threadIdx.x;       // node*256 + j
    int n = id >> 8, j = id & 255;
    float a = b1[j], b = 0.f;
#pragma unroll
    for (int k = 0; k < 10; ++k) {
        float xv = x[n * 10 + k];
        a += xv * W1[k * HID + j];
        b += xv * W1[(10 + k) * HID + j];
    }
    hA[id] = (__bf16)a;
    hB[id] = (__bf16)b;
}

// ------------------------------------------------- node transform GEMM (K=256, WMMA)
__global__ void __launch_bounds__(32)
k_nodegemm(const __bf16* __restrict__ hin, const __bf16* __restrict__ wp,
           const float* __restrict__ bias, __bf16* __restrict__ out) {
    int lane = threadIdx.x;
    int m0 = blockIdx.x * 16;      // node tile
    int nt = blockIdx.y;           // 16-wide output tile
    int m = lane & 15, ko = (lane >> 4) * 8;
    const __bf16* arow = hin + (size_t)(m0 + m) * HID;
    v8f acc = {};
#pragma unroll
    for (int kc = 0; kc < 8; ++kc) {
        Frag a, b;
        a.u[0] = *(const uint4*)(arow + kc * 32 + ko);
        a.u[1] = *(const uint4*)(arow + kc * 32 + 16 + ko);
        const uint4* bp = (const uint4*)wp + ((size_t)(kc * 16 + nt) * 32 + lane) * 2;
        b.u[0] = bp[0]; b.u[1] = bp[1];
        acc = __builtin_amdgcn_wmma_f32_16x16x32_bf16(false, a.v, false, b.v,
                                                      (short)0, acc, false, false);
    }
    int n  = nt * 16 + (lane & 15);
    int mb = (lane >> 4) * 8;
    float bv = bias ? bias[n] : 0.f;
#pragma unroll
    for (int i = 0; i < 8; ++i)
        out[(size_t)(m0 + mb + i) * HID + n] = (__bf16)(acc[i] + bv);
}

// ------------------------------------------------- edge kernel: gather -> 2x WMMA GEMM -> atomic scatter
// One wave handles EPT=32 edges (two 16-row M-tiles); B fragments are reused across both
// M-tiles, halving L2 weight traffic vs a 16-edge tile.
__global__ void __launch_bounds__(32)
k_edge(const int* __restrict__ srcI, const int* __restrict__ dstI,
       const __bf16* __restrict__ hA, const __bf16* __restrict__ hB,
       const __bf16* __restrict__ w2p, const float* __restrict__ b2,
       const __bf16* __restrict__ w3p, const float* __restrict__ b3,
       float* __restrict__ accN) {
    __shared__ __align__(16) __bf16 t[EPT * TROW];
    __shared__ __align__(16) __bf16 u[EPT * TROW];
    __shared__ int sidx[2 * EPT];   // [0..31]=dst, [32..63]=src

    int lane = threadIdx.x;
    int e0 = blockIdx.x * EPT;
    sidx[lane]       = dstI[e0 + lane];
    sidx[EPT + lane] = srcI[e0 + lane];
    __syncthreads();

    // gather: t = relu(A[dst] + B[src]) , 32 rows x 256 bf16; row == loop counter (uniform idx)
#pragma unroll
    for (int r = 0; r < EPT; ++r) {
        uint4 ua = *((const uint4*)(hA + (size_t)sidx[r]       * HID) + lane);
        uint4 ub = *((const uint4*)(hB + (size_t)sidx[EPT + r] * HID) + lane);
        const __bf16* pa = (const __bf16*)&ua;
        const __bf16* pb = (const __bf16*)&ub;
        __align__(16) __bf16 res[8];
#pragma unroll
        for (int j = 0; j < 8; ++j)
            res[j] = (__bf16)fmaxf((float)pa[j] + (float)pb[j], 0.f);
        *(uint4*)&t[r * TROW + lane * 8] = *(const uint4*)res;
    }
    __syncthreads();

    int m  = lane & 15, ko = (lane >> 4) * 8;
    int nn = lane & 15, mb = (lane >> 4) * 8;
    v8f z = {};

    // GEMM1: u = relu(t @ W2 + b2)
#pragma unroll 1
    for (int nt = 0; nt < 16; ++nt) {
        v8f acc0 = z, acc1 = z;
#pragma unroll
        for (int kc = 0; kc < 8; ++kc) {
            Frag b;
            const uint4* bp = (const uint4*)w2p + ((size_t)(kc * 16 + nt) * 32 + lane) * 2;
            b.u[0] = bp[0]; b.u[1] = bp[1];
            Frag a0, a1;
            a0.u[0] = *(const uint4*)&t[m * TROW + kc * 32 + ko];
            a0.u[1] = *(const uint4*)&t[m * TROW + kc * 32 + 16 + ko];
            a1.u[0] = *(const uint4*)&t[(16 + m) * TROW + kc * 32 + ko];
            a1.u[1] = *(const uint4*)&t[(16 + m) * TROW + kc * 32 + 16 + ko];
            acc0 = __builtin_amdgcn_wmma_f32_16x16x32_bf16(false, a0.v, false, b.v,
                                                           (short)0, acc0, false, false);
            acc1 = __builtin_amdgcn_wmma_f32_16x16x32_bf16(false, a1.v, false, b.v,
                                                           (short)0, acc1, false, false);
        }
        int n = nt * 16 + nn;
        float bv = b2[n];
#pragma unroll
        for (int i = 0; i < 8; ++i) {
            u[(mb + i) * TROW + n]      = (__bf16)fmaxf(acc0[i] + bv, 0.f);
            u[(16 + mb + i) * TROW + n] = (__bf16)fmaxf(acc1[i] + bv, 0.f);
        }
    }
    __syncthreads();

    // GEMM2: v = u @ W3 + b3 ; atomic scatter into accN[dst]
#pragma unroll 1
    for (int nt = 0; nt < 16; ++nt) {
        v8f acc0 = z, acc1 = z;
#pragma unroll
        for (int kc = 0; kc < 8; ++kc) {
            Frag b;
            const uint4* bp = (const uint4*)w3p + ((size_t)(kc * 16 + nt) * 32 + lane) * 2;
            b.u[0] = bp[0]; b.u[1] = bp[1];
            Frag a0, a1;
            a0.u[0] = *(const uint4*)&u[m * TROW + kc * 32 + ko];
            a0.u[1] = *(const uint4*)&u[m * TROW + kc * 32 + 16 + ko];
            a1.u[0] = *(const uint4*)&u[(16 + m) * TROW + kc * 32 + ko];
            a1.u[1] = *(const uint4*)&u[(16 + m) * TROW + kc * 32 + 16 + ko];
            acc0 = __builtin_amdgcn_wmma_f32_16x16x32_bf16(false, a0.v, false, b.v,
                                                           (short)0, acc0, false, false);
            acc1 = __builtin_amdgcn_wmma_f32_16x16x32_bf16(false, a1.v, false, b.v,
                                                           (short)0, acc1, false, false);
        }
        int n = nt * 16 + nn;
        float bv = b3[n];
#pragma unroll
        for (int i = 0; i < 8; ++i) {
            atomicAdd(&accN[(size_t)sidx[mb + i] * HID + n],      acc0[i] + bv);
            atomicAdd(&accN[(size_t)sidx[16 + mb + i] * HID + n], acc1[i] + bv);
        }
    }
}

// ------------------------------------------------- inter-layer: hcur = bf16(relu(acc)); acc = 0
__global__ void k_finalize(float* __restrict__ acc, __bf16* __restrict__ hcur) {
    int id = blockIdx.x * 256 + threadIdx.x;
    float v = acc[id];
    acc[id] = 0.f;
    hcur[id] = (__bf16)fmaxf(v, 0.f);
}

// ------------------------------------------------- mean pool (sums + counts via atomics)
__global__ void k_pool(const float* __restrict__ acc, const int* __restrict__ batch,
                       float* __restrict__ pool, int* __restrict__ cnt) {
    int id = blockIdx.x * 256 + threadIdx.x;
    int n = id >> 8, j = id & 255;
    int g = batch[n];
    atomicAdd(&pool[g * HID + j], acc[id]);
    if (j == 0) atomicAdd(&cnt[g], 1);
}

// ------------------------------------------------- classifier head (64x101, tiny)
__global__ void k_head(const float* __restrict__ pool, const int* __restrict__ cnt,
                       const float* __restrict__ Wout, const float* __restrict__ bout,
                       float* __restrict__ out) {
    int g = blockIdx.x, c = threadIdx.x;
    if (c >= NCLS) return;
    float inv = 1.f / fmaxf((float)cnt[g], 1.f);
    float s = bout[c];
    for (int k = 0; k < HID; ++k)
        s += pool[g * HID + k] * inv * Wout[k * NCLS + c];
    out[g * NCLS + c] = s;
}

// =================================================================== launch
extern "C" void kernel_launch(void* const* d_in, const int* in_sizes, int n_in,
                              void* d_out, int out_size, void* d_ws, size_t ws_size,
                              hipStream_t stream) {
    (void)in_sizes; (void)n_in; (void)out_size; (void)ws_size;
    const float* x     = (const float*)d_in[0];
    const int*   ei    = (const int*)d_in[1];
    const int*   srcI  = ei;
    const int*   dstI  = ei + N_EDGES;
    const int*   batch = (const int*)d_in[2];
    const float* W[4][3]; const float* B[4][3];
    for (int l = 0; l < 4; ++l)
        for (int m = 0; m < 3; ++m) {
            W[l][m] = (const float*)d_in[3 + l * 6 + m * 2];
            B[l][m] = (const float*)d_in[3 + l * 6 + m * 2 + 1];
        }
    const float* Wout = (const float*)d_in[27];
    const float* bout = (const float*)d_in[28];
    float* out = (float*)d_out;

    // workspace carve-up
    char* ws = (char*)d_ws;
    size_t off = 0;
    auto carve = [&](size_t bytes) {
        void* p = ws + off;
        off += (bytes + 255) & ~(size_t)255;
        return p;
    };
    __bf16* hcur = (__bf16*)carve((size_t)N_NODES * HID * 2);
    __bf16* hA   = (__bf16*)carve((size_t)N_NODES * HID * 2);
    __bf16* hB   = (__bf16*)carve((size_t)N_NODES * HID * 2);
    float*  acc  = (float*)carve((size_t)N_NODES * HID * 4);
    const size_t PS = (size_t)HID * HID;           // packed matrix elements
    __bf16* packs = (__bf16*)carve(14 * PS * 2);
    float*  pool  = (float*)carve((size_t)NGRAPH * HID * 4);
    int*    cnt   = (int*)carve((size_t)NGRAPH * 4);

    __bf16 *w1tp[4], *w1bp[4], *w2p[4], *w3p[4];
    __bf16* p = packs;
    w2p[0] = p; p += PS; w3p[0] = p; p += PS;
    for (int l = 1; l < 4; ++l) {
        w1tp[l] = p; p += PS;
        w1bp[l] = p; p += PS;
        w2p[l]  = p; p += PS;
        w3p[l]  = p; p += PS;
    }

    const int totalP = HID * HID;   // K=256 packed element count

    k_init<<<N_NODES, 256, 0, stream>>>(acc, pool, cnt);

    k_pack<<<totalP / 256, 256, 0, stream>>>(W[0][1], 0, w2p[0], totalP);
    k_pack<<<totalP / 256, 256, 0, stream>>>(W[0][2], 0, w3p[0], totalP);
    for (int l = 1; l < 4; ++l) {
        k_pack<<<totalP / 256, 256, 0, stream>>>(W[l][0], 0,   w1tp[l], totalP);
        k_pack<<<totalP / 256, 256, 0, stream>>>(W[l][0], 256, w1bp[l], totalP);
        k_pack<<<totalP / 256, 256, 0, stream>>>(W[l][1], 0,   w2p[l],  totalP);
        k_pack<<<totalP / 256, 256, 0, stream>>>(W[l][2], 0,   w3p[l],  totalP);
    }

    // ---- layer 0
    k_node0<<<N_NODES, 256, 0, stream>>>(x, W[0][0], B[0][0], hA, hB);
    k_edge<<<N_EDGES / EPT, 32, 0, stream>>>(srcI, dstI, hA, hB,
                                             w2p[0], B[0][1], w3p[0], B[0][2], acc);
    k_finalize<<<N_NODES, 256, 0, stream>>>(acc, hcur);

    // ---- layers 1..3
    for (int l = 1; l < 4; ++l) {
        dim3 g(N_NODES / 16, 16);
        k_nodegemm<<<g, 32, 0, stream>>>(hcur, w1tp[l], B[l][0], hA);
        k_nodegemm<<<g, 32, 0, stream>>>(hcur, w1bp[l], nullptr, hB);
        k_edge<<<N_EDGES / EPT, 32, 0, stream>>>(srcI, dstI, hA, hB,
                                                 w2p[l], B[l][1], w3p[l], B[l][2], acc);
        if (l < 3) k_finalize<<<N_NODES, 256, 0, stream>>>(acc, hcur);
    }

    // ---- pooling + head
    k_pool<<<N_NODES, 256, 0, stream>>>(acc, batch, pool, cnt);
    k_head<<<NGRAPH, 128, 0, stream>>>(pool, cnt, Wout, bout, out);
}
